// whole_network_84688165143287
// MI455X (gfx1250) — matrix operand order repaired and checked by reading, not compiled, not run on hardware
//
#include <hip/hip_runtime.h>

// Problem constants (match reference)
#define BB   16
#define RR   400
#define TT   400
#define HH   8
#define QQ   64
#define HQv  512
#define NGCN 8

typedef __attribute__((ext_vector_type(16))) __bf16 v16bf;
typedef __attribute__((ext_vector_type(8)))  float  v8f;

union BfVec { v16bf v; unsigned short u[16]; };
union U4v   { uint4 q; unsigned short u[8]; };

__device__ __forceinline__ unsigned short f2bf(float f) {
  unsigned u = __builtin_bit_cast(unsigned, f);
  unsigned r = 0x7fffu + ((u >> 16) & 1u);
  return (unsigned short)((u + r) >> 16);
}
__device__ __forceinline__ float bf2f(unsigned short s) {
  unsigned u = ((unsigned)s) << 16;
  return __builtin_bit_cast(float, u);
}

// ---------------------------------------------------------------------------
// Tile stagers for the WMMA GEMM. Guards are template parameters so the
// steady-state path (interior block, full K step) is completely unpredicated:
// pure global_load_b128 -> ds_store. K/N remainders are multiples of 8 at
// every call site, so one all-or-nothing check per 16B group is exact.
// ---------------------------------------------------------------------------
template<bool GM, bool GK>
__device__ __forceinline__ void stage_A(unsigned short (&lA)[128][40],
                                        const unsigned short* __restrict__ A,
                                        int bm, int k0, int M, int K, int lda,
                                        int tid) {
  const uint4 zero4 = {0u, 0u, 0u, 0u};
  #pragma unroll
  for (int t = 0; t < 2; ++t) {
    int g   = tid + t * 256;
    int row = g >> 2;               // 0..127
    int c8  = (g & 3) << 3;         // 0,8,16,24
    int gm  = bm + row, gk = k0 + c8;
    bool ok = true;
    if (GM) ok = ok && (gm < M);
    if (GK) ok = ok && (gk < K);
    uint4 v = ok ? *reinterpret_cast<const uint4*>(A + (long long)gm * lda + gk)
                 : zero4;
    *reinterpret_cast<uint4*>(&lA[row][c8]) = v;
  }
}

template<bool GN, bool GK>
__device__ __forceinline__ void stage_B(unsigned short (&lB)[128][40],
                                        const unsigned short* __restrict__ Bm,
                                        int bn, int k0, int N, int K, int ldb,
                                        int tid) {
  const uint4 zero4 = {0u, 0u, 0u, 0u};
  #pragma unroll
  for (int t = 0; t < 2; ++t) {
    int g  = tid + t * 256;
    int k  = g >> 4;                // 0..31
    int n8 = (g & 15) << 3;         // 0..120
    int gk = k0 + k, gn = bn + n8;
    bool ok = true;
    if (GN) ok = ok && (gn < N);
    if (GK) ok = ok && (gk < K);
    U4v v; v.q = ok ? *reinterpret_cast<const uint4*>(Bm + (long long)gk * ldb + gn)
                    : zero4;
    #pragma unroll
    for (int e = 0; e < 8; ++e) lB[n8 + e][k] = v.u[e];   // transpose to [n][k]
  }
}

// One 32-deep K step: assemble fragments per ISA 7.12.2 layouts and issue
// the 2x4 grid of v_wmma_f32_16x16x32_bf16.
__device__ __forceinline__ void mma_step(const unsigned short (&lA)[128][40],
                                         const unsigned short (&lB)[128][40],
                                         int wm, int wn, int lhalf, int l16,
                                         v8f (&acc)[2][4]) {
  // A fragment (16x32): lanes 0-15 hold K 0-7 / 16-23 of their row,
  // lanes 16-31 hold K 8-15 / 24-31.
  BfVec af[2];
  #pragma unroll
  for (int i = 0; i < 2; ++i) {
    int row = wm + i * 16 + l16;
    int kb  = lhalf * 8;
    #pragma unroll
    for (int e = 0; e < 8; ++e) {
      af[i].u[e]     = lA[row][kb + e];
      af[i].u[8 + e] = lA[row][16 + kb + e];
    }
  }
  // B fragment (32x16): lanes 0-15 hold K 0-7 / 16-23 of their column,
  // lanes 16-31 hold K 8-15 / 24-31 (same order as A).
  BfVec bfm[4];
  #pragma unroll
  for (int j = 0; j < 4; ++j) {
    int col = wn + j * 16 + l16;
    int kb  = lhalf * 8;
    #pragma unroll
    for (int e = 0; e < 8; ++e) {
      bfm[j].u[e]     = lB[col][kb + e];
      bfm[j].u[8 + e] = lB[col][16 + kb + e];
    }
  }
  #pragma unroll
  for (int i = 0; i < 2; ++i)
    #pragma unroll
    for (int j = 0; j < 4; ++j)
      acc[i][j] = __builtin_amdgcn_wmma_f32_16x16x32_bf16(
          false, af[i].v, false, bfm[j].v, (short)0, acc[i][j], false, false);
}

// ---------------------------------------------------------------------------
// Generic batched bf16 WMMA GEMM:  C = epilogue( scale * A@B + bias ) [+res]
// A: [M,K] bf16 row-major (ushort), B: [K,N] bf16 row-major.
// 256 threads = 8 wave32 waves; block tile 128x128; wave tile 32x64;
// double-buffered LDS, one barrier per 32-deep K step.
//
// REQUIREMENTS (all call sites satisfy them):
//   K % 8 == 0, N % 8 == 0, lda % 8 == 0, ldb % 8 == 0,
//   A/B base pointers 16B aligned, batch strides keep 16B alignment.
// ---------------------------------------------------------------------------
__global__ void __launch_bounds__(256)
gemm_bf16(const unsigned short* __restrict__ A, long long sAb, int lda,
          const unsigned short* __restrict__ Bm, long long sBb, int ldb,
          float* Cf, long long sCfb,
          unsigned short* Cb, long long sCbb,
          int ldc,
          const float* bias, long long sBiasb,
          const float* res, long long sResb,
          int M, int N, int K, float scale, int do_relu)
{
  // Row pitch 40 halfs = 80 B: 16B-aligned vector stores; 20 banks/row
  // spreads the 16 row-lanes of a fragment over distinct banks.
  __shared__ unsigned short lA[2][128][40];
  __shared__ unsigned short lB[2][128][40];

  const int bz = blockIdx.z;
  A  += bz * sAb;
  Bm += bz * sBb;
  if (Cf)   Cf   += bz * sCfb;
  if (Cb)   Cb   += bz * sCbb;
  if (bias) bias += bz * sBiasb;
  if (res)  res  += bz * sResb;

  const int bm   = blockIdx.y * 128;
  const int bn   = blockIdx.x * 128;
  const int tid  = threadIdx.x;
  const int w    = tid >> 5;
  const int lane = tid & 31;
  const int wm   = (w & 3) * 32;   // wave M offset in block tile
  const int wn   = (w >> 2) * 64;  // wave N offset in block tile
  const int lhalf = lane >> 4;     // 0: lanes 0-15, 1: lanes 16-31
  const int l16   = lane & 15;

  v8f acc[2][4];
  {
    v8f z = {0.f, 0.f, 0.f, 0.f, 0.f, 0.f, 0.f, 0.f};
    for (int i = 0; i < 2; ++i)
      for (int j = 0; j < 4; ++j)
        acc[i][j] = z;
  }

  const bool mn_ok  = (bm + 128 <= M) && (bn + 128 <= N);
  const int  nsteps = (K + 31) >> 5;

  // Uniform per-step dispatch into the right guard specialization.
  auto stage_any = [&](int buf, int k0) {
    bool gk = (k0 + 32 > K);
    if (mn_ok) {
      if (gk) { stage_A<false, true >(lA[buf], A, bm, k0, M, K, lda, tid);
                stage_B<false, true >(lB[buf], Bm, bn, k0, N, K, ldb, tid); }
      else    { stage_A<false, false>(lA[buf], A, bm, k0, M, K, lda, tid);
                stage_B<false, false>(lB[buf], Bm, bn, k0, N, K, ldb, tid); }
    } else {
      if (gk) { stage_A<true,  true >(lA[buf], A, bm, k0, M, K, lda, tid);
                stage_B<true,  true >(lB[buf], Bm, bn, k0, N, K, ldb, tid); }
      else    { stage_A<true,  false>(lA[buf], A, bm, k0, M, K, lda, tid);
                stage_B<true,  false>(lB[buf], Bm, bn, k0, N, K, ldb, tid); }
    }
  };

  stage_any(0, 0);
  int p = 0;
  for (int s = 0; s < nsteps; ++s) {
    __syncthreads();
    if (s + 1 < nsteps) {
      if (tid == 0 && (s + 2) * 32 < K)   // prefetch step s+2 A slab
        __builtin_prefetch(A + (long long)bm * lda + (s + 2) * 32, 0, 1);
      stage_any(p ^ 1, (s + 1) * 32);
    }
    mma_step(lA[p], lB[p], wm, wn, lhalf, l16, acc);
    p ^= 1;
  }

  // Epilogue. C/D layout: VGPR r -> M = lhalf*8 + r, N = l16 (ISA 7.12.2).
  if (mn_ok) {
    float bv[4];
    #pragma unroll
    for (int j = 0; j < 4; ++j)
      bv[j] = bias ? bias[bn + wn + j * 16 + l16] : 0.f;
    #pragma unroll
    for (int i = 0; i < 2; ++i) {
      #pragma unroll
      for (int j = 0; j < 4; ++j) {
        #pragma unroll
        for (int r = 0; r < 8; ++r) {
          long long m = bm + wm + i * 16 + lhalf * 8 + r;
          long long n = bn + wn + j * 16 + l16;
          float v = acc[i][j][r] * scale + bv[j];
          if (do_relu) v = v > 0.f ? v : 0.f;
          if (res)     v += res[m * ldc + n];
          if (Cf) Cf[m * ldc + n] = v;
          if (Cb) Cb[m * ldc + n] = f2bf(v);
        }
      }
    }
  } else {
    #pragma unroll
    for (int i = 0; i < 2; ++i) {
      #pragma unroll
      for (int j = 0; j < 4; ++j) {
        #pragma unroll
        for (int r = 0; r < 8; ++r) {
          int m = bm + wm + i * 16 + lhalf * 8 + r;
          int n = bn + wn + j * 16 + l16;
          if (m < M && n < N) {
            float v = acc[i][j][r] * scale;
            if (bias)    v += bias[n];
            if (do_relu) v  = v > 0.f ? v : 0.f;
            if (res)     v += res[(long long)m * ldc + n];
            if (Cf) Cf[(long long)m * ldc + n] = v;
            if (Cb) Cb[(long long)m * ldc + n] = f2bf(v);
          }
        }
      }
    }
  }
}

// ---------------------------------------------------------------------------
// Elementwise f32 -> bf16
// ---------------------------------------------------------------------------
__global__ void cvt_bf16_kernel(const float* __restrict__ src,
                                unsigned short* __restrict__ dst,
                                long long n) {
  long long i = blockIdx.x * 256LL + threadIdx.x;
  long long stride = (long long)gridDim.x * 256LL;
  for (; i < n; i += stride) dst[i] = f2bf(src[i]);
}

// qk [B,R,H,Q] f32 -> qkA bf16 [B,H,R,Q] and qkT bf16 [B,H,Q,R]
__global__ void repack_qk_kernel(const float* __restrict__ qk,
                                 unsigned short* __restrict__ qkA,
                                 unsigned short* __restrict__ qkT,
                                 long long n) {
  long long i = blockIdx.x * 256LL + threadIdx.x;
  long long stride = (long long)gridDim.x * 256LL;
  for (; i < n; i += stride) {
    long long q = i % QQ;
    long long h = (i / QQ) % HH;
    long long r = (i / (QQ * HH)) % RR;
    long long b = i / (QQ * HH * RR);
    unsigned short v = f2bf(qk[i]);
    long long bh = b * HH + h;
    qkA[(bh * RR + r) * QQ + q] = v;
    qkT[(bh * QQ + q) * RR + r] = v;
  }
}

// ---------------------------------------------------------------------------
// Row-wise relu + softmax, in place. One 256-thread block per row of 400.
// ---------------------------------------------------------------------------
__global__ void __launch_bounds__(256)
relu_softmax_rows(float* __restrict__ rel) {
  __shared__ float red[256];
  float* p = rel + (long long)blockIdx.x * RR;
  int tid = threadIdx.x;

  float lmax = -1e30f;
  for (int i = tid; i < RR; i += 256) {
    float v = p[i];
    v = v > 0.f ? v : 0.f;
    p[i] = v;
    lmax = fmaxf(lmax, v);
  }
  red[tid] = lmax;
  __syncthreads();
  for (int s = 128; s > 0; s >>= 1) {
    if (tid < s) red[tid] = fmaxf(red[tid], red[tid + s]);
    __syncthreads();
  }
  float mx = red[0];
  __syncthreads();

  float lsum = 0.f;
  for (int i = tid; i < RR; i += 256) {
    float e = __expf(p[i] - mx);
    p[i] = e;
    lsum += e;
  }
  red[tid] = lsum;
  __syncthreads();
  for (int s = 128; s > 0; s >>= 1) {
    if (tid < s) red[tid] += red[tid + s];
    __syncthreads();
  }
  float inv = 1.f / red[0];
  for (int i = tid; i < RR; i += 256) p[i] *= inv;
}

// ---------------------------------------------------------------------------
// SE gate. Post-softmax rows each sum to 1, so gap[b,h] == 1/R exactly for
// all b,h: the gate is input-independent -> compute gate[H] once.
// ---------------------------------------------------------------------------
__global__ void se_gate_kernel(const float* __restrict__ W1,
                               const float* __restrict__ b1,
                               const float* __restrict__ W2,
                               const float* __restrict__ b2,
                               float* __restrict__ gate) {
  if (threadIdx.x == 0 && blockIdx.x == 0) {
    const float gap = 1.0f / (float)RR;
    float hid[HH / 2];
    for (int j = 0; j < HH / 2; ++j) {
      float s = b1[j];
      for (int h = 0; h < HH; ++h) s += gap * W1[h * (HH / 2) + j];
      hid[j] = s > 0.f ? s : 0.f;
    }
    for (int h = 0; h < HH; ++h) {
      float s = b2[h];
      for (int j = 0; j < HH / 2; ++j) s += hid[j] * W2[j * HH + h];
      gate[h] = 1.f / (1.f + __expf(-s));
    }
  }
}

// ---------------------------------------------------------------------------
// Dilated mask -> sparse adjacency. Stable ascending argsort semantics:
// rank(j) = #{i: v_i < v_j} + #{i<j: v_i == v_j}. Selected columns are the
// ranks of original columns {0,16..20} plus the diagonal; <=7 nnz per row.
// Gate (positive) folded into values (doesn't change ranks).
// ---------------------------------------------------------------------------
__global__ void __launch_bounds__(256)
build_adj_kernel(const float* __restrict__ rel, const float* __restrict__ gate,
                 int* __restrict__ cols, float* __restrict__ vals) {
  __shared__ float rowv[RR];
  __shared__ int cnt[6];
  int rid = blockIdx.x;              // rid = (b*H + h)*R + r
  int r   = rid % RR;
  int h   = (rid / RR) % HH;
  const float* p = rel + (long long)rid * RR;
  int tid = threadIdx.x;

  for (int i = tid; i < RR; i += 256) rowv[i] = p[i];
  if (tid < 6) cnt[tid] = 0;
  __syncthreads();

  const int tgt[6] = {0, 16, 17, 18, 19, 20};
  for (int i = tid; i < RR; i += 256) {
    float v = rowv[i];
    #pragma unroll
    for (int jj = 0; jj < 6; ++jj) {
      int j = tgt[jj];
      float tv = rowv[j];
      if (v < tv || (v == tv && i < j)) atomicAdd(&cnt[jj], 1);
    }
  }
  __syncthreads();

  if (tid == 0) {
    float g = gate[h];
    int c[8]; float wv[8]; int n = 0;
    #pragma unroll
    for (int jj = 0; jj < 6; ++jj) {
      int k = cnt[jj];
      c[n] = k; wv[n] = rowv[k] * g; ++n;
    }
    bool hasDiag = false;
    for (int e = 0; e < 6; ++e) if (c[e] == r) hasDiag = true;
    if (!hasDiag) { c[n] = r; wv[n] = rowv[r] * g; ++n; }
    for (; n < 8; ++n) { c[n] = 0; wv[n] = 0.f; }
    for (int e = 0; e < 8; ++e) {
      cols[(long long)rid * 8 + e] = c[e];
      vals[(long long)rid * 8 + e] = wv[e];
    }
  }
}

// feats[b,h,r,t] = x[b,r,t]  (float4 vectorized; TT % 4 == 0)
__global__ void init_feats_kernel(const float* __restrict__ x,
                                  float* __restrict__ feats, long long total4) {
  const long long T4 = TT / 4;
  long long i = blockIdx.x * 256LL + threadIdx.x;
  long long stride = (long long)gridDim.x * 256LL;
  for (; i < total4; i += stride) {
    long long t4  = i % T4;
    long long rem = i / T4;
    long long r   = rem % RR;
    long long b   = (rem / RR) / HH;
    float4 v = *reinterpret_cast<const float4*>(x + (b * RR + r) * TT + t4 * 4);
    *reinterpret_cast<float4*>(feats + i * 4) = v;
  }
}

// msg[bh,r,:] = sum_e vals[e] * feats[bh, cols[e], :]  (bf16 out for WMMA A)
__global__ void __launch_bounds__(128)
spmm_msg_kernel(const int* __restrict__ cols, const float* __restrict__ vals,
                const float* __restrict__ feats, unsigned short* __restrict__ msg) {
  __shared__ int   sc[8];
  __shared__ float sv[8];
  int rid = blockIdx.x;                 // bh*R + r
  long long bh = rid / RR;
  if (threadIdx.x < 8) {
    sc[threadIdx.x] = cols[(long long)rid * 8 + threadIdx.x];
    sv[threadIdx.x] = vals[(long long)rid * 8 + threadIdx.x];
  }
  __syncthreads();
  const float* fb = feats + bh * (long long)RR * TT;
  unsigned short* mrow = msg + (long long)rid * TT;
  for (int t4 = threadIdx.x; t4 < TT / 4; t4 += 128) {
    float4 s = {0.f, 0.f, 0.f, 0.f};
    #pragma unroll
    for (int e = 0; e < 8; ++e) {
      float4 f = *reinterpret_cast<const float4*>(fb + (long long)sc[e] * TT + t4 * 4);
      s.x += sv[e] * f.x; s.y += sv[e] * f.y;
      s.z += sv[e] * f.z; s.w += sv[e] * f.w;
    }
    union { unsigned short u[4]; uint2 q; } o;
    o.u[0] = f2bf(s.x); o.u[1] = f2bf(s.y);
    o.u[2] = f2bf(s.z); o.u[3] = f2bf(s.w);
    *reinterpret_cast<uint2*>(mrow + t4 * 4) = o.q;
  }
}

// fused[b,r,t] = sum_h feats[b,h,r,t] * fw[h] + fb   (bf16 out for MLP)
__global__ void fuse_heads_kernel(const float* __restrict__ feats,
                                  const float* __restrict__ fw,
                                  const float* __restrict__ fbp,
                                  unsigned short* __restrict__ fused_bf,
                                  long long total4) {
  const long long T4 = TT / 4;
  long long i = blockIdx.x * 256LL + threadIdx.x;
  long long stride = (long long)gridDim.x * 256LL;
  float w[HH];
  #pragma unroll
  for (int h = 0; h < HH; ++h) w[h] = fw[h];
  float b0 = fbp[0];
  for (; i < total4; i += stride) {
    long long t4  = i % T4;
    long long rem = i / T4;
    long long r   = rem % RR;
    long long b   = rem / RR;
    float4 s = {b0, b0, b0, b0};
    #pragma unroll
    for (int h = 0; h < HH; ++h) {
      float4 f = *reinterpret_cast<const float4*>(
          feats + ((b * HH + h) * RR + r) * (long long)TT + t4 * 4);
      s.x += w[h] * f.x; s.y += w[h] * f.y;
      s.z += w[h] * f.z; s.w += w[h] * f.w;
    }
    union { unsigned short u[4]; uint2 q; } o;
    o.u[0] = f2bf(s.x); o.u[1] = f2bf(s.y);
    o.u[2] = f2bf(s.z); o.u[3] = f2bf(s.w);
    *reinterpret_cast<uint2*>(fused_bf + i * 4) = o.q;
  }
}

// out[m, c] = relu(h3[m,:] @ W4[:,c] + b4[c]),  c < 5
__global__ void final_fc_kernel(const unsigned short* __restrict__ h3,
                                const float* __restrict__ W4,
                                const float* __restrict__ b4,
                                float* __restrict__ out) {
  int idx = blockIdx.x * 256 + threadIdx.x;
  if (idx >= BB * RR * 5) return;
  int c = idx % 5, m = idx / 5;
  float s = b4[c];
  #pragma unroll 8
  for (int k = 0; k < 64; ++k) s += bf2f(h3[m * 64 + k]) * W4[k * 5 + c];
  out[idx] = s > 0.f ? s : 0.f;
}

// ---------------------------------------------------------------------------
extern "C" void kernel_launch(void* const* d_in, const int* in_sizes, int n_in,
                              void* d_out, int out_size, void* d_ws, size_t ws_size,
                              hipStream_t stream) {
  const float* x      = (const float*)d_in[0];
  const float* Wk     = (const float*)d_in[1];
  const float* bk     = (const float*)d_in[2];
  const float* se_W1  = (const float*)d_in[3];
  const float* se_b1  = (const float*)d_in[4];
  const float* se_W2  = (const float*)d_in[5];
  const float* se_b2  = (const float*)d_in[6];
  const float* gcn_W  = (const float*)d_in[7];
  const float* gcn_b  = (const float*)d_in[8];
  const float* fuse_w = (const float*)d_in[9];
  const float* fuse_b = (const float*)d_in[10];
  const float* mlp_W1 = (const float*)d_in[11];
  const float* mlp_b1 = (const float*)d_in[12];
  const float* mlp_W2 = (const float*)d_in[13];
  const float* mlp_b2 = (const float*)d_in[14];
  const float* mlp_W3 = (const float*)d_in[15];
  const float* mlp_b3 = (const float*)d_in[16];
  const float* mlp_W4 = (const float*)d_in[17];
  const float* mlp_b4 = (const float*)d_in[18];
  float* out = (float*)d_out;

  // ---- workspace bump allocator (256B aligned slices) ----
  char* ws = (char*)d_ws;
  size_t off = 0;
  auto alloc = [&](size_t bytes) -> void* {
    void* p = ws + off;
    off += (bytes + 255) & ~(size_t)255;
    return p;
  };
  const long long BHRT = (long long)BB * HH * RR * TT;   // 20,480,000
  float*          feats   = (float*)         alloc(BHRT * 4);
  float*          rel     = (float*)         alloc((long long)BB * HH * RR * RR * 4);
  unsigned short* msg     = (unsigned short*)alloc(BHRT * 2);
  float*          qk      = (float*)         alloc((long long)BB * RR * HQv * 4);
  unsigned short* qkA     = (unsigned short*)alloc((long long)BB * RR * HQv * 2);
  unsigned short* qkT     = (unsigned short*)alloc((long long)BB * RR * HQv * 2);
  unsigned short* x_bf    = (unsigned short*)alloc((long long)BB * RR * TT * 2);
  unsigned short* Wk_bf   = (unsigned short*)alloc((long long)TT * HQv * 2);
  unsigned short* gW_bf   = (unsigned short*)alloc((long long)NGCN * TT * TT * 2);
  unsigned short* W1_bf   = (unsigned short*)alloc((long long)TT * 256 * 2);
  unsigned short* W2_bf   = (unsigned short*)alloc(256LL * 128 * 2);
  unsigned short* W3_bf   = (unsigned short*)alloc(128LL * 64 * 2);
  unsigned short* fusedbf = (unsigned short*)alloc((long long)BB * RR * TT * 2);
  unsigned short* h1      = (unsigned short*)alloc((long long)BB * RR * 256 * 2);
  unsigned short* h2      = (unsigned short*)alloc((long long)BB * RR * 128 * 2);
  unsigned short* h3      = (unsigned short*)alloc((long long)BB * RR * 64 * 2);
  int*            acols   = (int*)           alloc((long long)BB * HH * RR * 8 * 4);
  float*          avals   = (float*)         alloc((long long)BB * HH * RR * 8 * 4);
  float*          gate    = (float*)         alloc(256);
  (void)ws_size; (void)in_sizes; (void)n_in; (void)out_size;

  const int ROWS = BB * HH * RR;     // 51200

  // ---- 1) bf16 conversions of weights / x ----
  cvt_bf16_kernel<<<1024, 256, 0, stream>>>(x,      x_bf,  (long long)BB * RR * TT);
  cvt_bf16_kernel<<<256,  256, 0, stream>>>(Wk,     Wk_bf, (long long)TT * HQv);
  cvt_bf16_kernel<<<1024, 256, 0, stream>>>(gcn_W,  gW_bf, (long long)NGCN * TT * TT);
  cvt_bf16_kernel<<<128,  256, 0, stream>>>(mlp_W1, W1_bf, (long long)TT * 256);
  cvt_bf16_kernel<<<64,   256, 0, stream>>>(mlp_W2, W2_bf, 256LL * 128);
  cvt_bf16_kernel<<<16,   256, 0, stream>>>(mlp_W3, W3_bf, 128LL * 64);

  // ---- 2) qk = x @ Wk + bk  : M=6400, K=400, N=512 ----
  {
    dim3 g((HQv + 127) / 128, (BB * RR + 127) / 128, 1);
    gemm_bf16<<<g, 256, 0, stream>>>(x_bf, 0, TT, Wk_bf, 0, HQv,
                                     qk, 0, nullptr, 0, HQv,
                                     bk, 0, nullptr, 0,
                                     BB * RR, HQv, TT, 1.f, 0);
  }
  repack_qk_kernel<<<2048, 256, 0, stream>>>(qk, qkA, qkT, (long long)BB * RR * HQv);

  // ---- 3) rel = (qk @ qk^T) / sqrt(Q) : batched x128, M=N=400, K=64 ----
  {
    dim3 g((RR + 127) / 128, (RR + 127) / 128, BB * HH);
    gemm_bf16<<<g, 256, 0, stream>>>(qkA, (long long)RR * QQ, QQ,
                                     qkT, (long long)QQ * RR, RR,
                                     rel, (long long)RR * RR, nullptr, 0, RR,
                                     nullptr, 0, nullptr, 0,
                                     RR, RR, QQ, 0.125f, 0);
  }

  // ---- 4) softmax(relu(rel)) per row; SE gate (constant gap = 1/R) ----
  relu_softmax_rows<<<ROWS, 256, 0, stream>>>(rel);
  se_gate_kernel<<<1, 32, 0, stream>>>(se_W1, se_b1, se_W2, se_b2, gate);

  // ---- 5) sparse adjacency from rank selection (+gate folded in) ----
  build_adj_kernel<<<ROWS, 256, 0, stream>>>(rel, gate, acols, avals);

  // ---- 6) GCN: feats = relu(adj@feats @ Wi + bi) + feats, 8 layers ----
  init_feats_kernel<<<4096, 256, 0, stream>>>(x, feats, BHRT / 4);
  for (int i = 0; i < NGCN; ++i) {
    spmm_msg_kernel<<<ROWS, 128, 0, stream>>>(acols, avals, feats, msg);
    dim3 g((TT + 127) / 128, (ROWS + 127) / 128, 1);
    gemm_bf16<<<g, 256, 0, stream>>>(msg, 0, TT,
                                     gW_bf + (long long)i * TT * TT, 0, TT,
                                     feats, 0, nullptr, 0, TT,
                                     gcn_b + (long long)i * TT, 0,
                                     feats, 0,
                                     ROWS, TT, TT, 1.f, 1);
  }

  // ---- 7) fuse heads -> bf16 activations for MLP ----
  fuse_heads_kernel<<<2048, 256, 0, stream>>>(feats, fuse_w, fuse_b, fusedbf,
                                              (long long)BB * RR * TT / 4);

  // ---- 8) MLP head (bf16 WMMA for the three big layers) ----
  {
    dim3 g((256 + 127) / 128, (BB * RR + 127) / 128, 1);
    gemm_bf16<<<g, 256, 0, stream>>>(fusedbf, 0, TT, W1_bf, 0, 256,
                                     nullptr, 0, h1, 0, 256,
                                     mlp_b1, 0, nullptr, 0,
                                     BB * RR, 256, TT, 1.f, 1);
  }
  {
    dim3 g(1, (BB * RR + 127) / 128, 1);
    gemm_bf16<<<g, 256, 0, stream>>>(h1, 0, 256, W2_bf, 0, 128,
                                     nullptr, 0, h2, 0, 128,
                                     mlp_b2, 0, nullptr, 0,
                                     BB * RR, 128, 256, 1.f, 1);
  }
  {
    dim3 g(1, (BB * RR + 127) / 128, 1);
    gemm_bf16<<<g, 256, 0, stream>>>(h2, 0, 128, W3_bf, 0, 64,
                                     nullptr, 0, h3, 0, 64,
                                     mlp_b3, 0, nullptr, 0,
                                     BB * RR, 64, 128, 1.f, 1);
  }
  final_fc_kernel<<<(BB * RR * 5 + 255) / 256, 256, 0, stream>>>(h3, mlp_W4, mlp_b4, out);
}
